// ResidualBlock_84327387890552
// MI455X (gfx1250) — matrix operand order repaired and checked by reading, not compiled, not run on hardware
//
#include <hip/hip_runtime.h>
#include <hip/hip_bf16.h>

// ---------------------------------------------------------------------------
// Two stacked ConvLSTM layers as implicit GEMM on bf16 WMMA (CDNA5 / gfx1250).
// gates = conv([x_t ; h_{t-1}], [Wx;Wh]) + b  ->  M=16384, N=192, K=9*96=864
// Weights for the current 3x3 tap are staged in LDS (double-buffered) by the
// Tensor Data Mover; B fragments then come from ds_load_b128 instead of
// latency-exposed global loads.
// ---------------------------------------------------------------------------

typedef __attribute__((ext_vector_type(16))) __bf16 bf16x16;
typedef __attribute__((ext_vector_type(8)))  float  f32x8;
typedef __attribute__((ext_vector_type(4)))  unsigned int u32x4;
typedef __attribute__((ext_vector_type(8)))  int i32x8;
typedef __attribute__((ext_vector_type(4)))  int i32x4;

#define B_   4
#define T_   16
#define HW_  4096        // 64*64
#define C_   48
#define CC_  96          // concat channels
#define NG_  192         // 4*48 gate channels
#define PIX_ (B_*HW_)    // 16384

#define ACT_LDS_BYTES  (4 * 64 * CC_ * 2)          // 49152
#define WTAP_ELEMS     (NG_ * CC_)                 // 18432 bf16 per 3x3 tap
#define WTAP_BYTES     (WTAP_ELEMS * 2)            // 36864
#define SMEM_BYTES     (ACT_LDS_BYTES + 2 * WTAP_BYTES)   // 122880

// ---------------------------------------------------------------------------
// TDM: DMA one 36 KB weight tap (contiguous) from global into LDS.
// Descriptor per CDNA5 ISA 8.3/8.4: 1-D copy, data_size = 8B,
// tile_dim0 = 4608 elements, type=2 ("image"), count=1.
// This toolchain's builtin takes 6 args (g0, g1, g2, g3, extra, cpol).
// ---------------------------------------------------------------------------
__device__ __forceinline__ void tdm_load_tap(unsigned ldsByteOff,
                                             const __hip_bfloat16* gsrc)
{
    unsigned long long ga = (unsigned long long)(uintptr_t)gsrc;
    u32x4 g0;
    g0[0] = 1u;                                       // count = 1 (user mode)
    g0[1] = ldsByteOff;                               // lds_addr
    g0[2] = (unsigned)(ga & 0xffffffffu);             // global_addr[31:0]
    g0[3] = (unsigned)((ga >> 32) & 0x01ffffffu)      // global_addr[56:32]
          | 0x80000000u;                              // type = 2
    i32x8 g1;
    g1[0] = 0x00030000;                 // workgroup_mask=0, data_size=3 (8B)
    g1[1] = (int)(4608u << 16);         // tensor_dim0[15:0] = 4608 (bit 48)
    g1[2] = (int)(1u << 16);            // tensor_dim1 = 1
    g1[3] = (int)(4608u << 16);         // tile_dim0 = 4608
    g1[4] = 1;                          // tile_dim1 = 1
    g1[5] = 4608;                       // tensor_dim0_stride = 4608
    g1[6] = 0;
    g1[7] = 0;
    i32x4 z4 = {0, 0, 0, 0};
    i32x8 z8 = {0, 0, 0, 0, 0, 0, 0, 0};
    __builtin_amdgcn_tensor_load_to_lds(g0, g1, z4, z4, z8, 0);
}

// ---------------------------------------------------------------------------
// Weight prep: Wb[layer][k9][n][c96] bf16, c<48 from Wx, c>=48 from Wh.
// Wx/Wh global layout: [3,3,C,4F] = [(k9*48 + c)*192 + n]
// ---------------------------------------------------------------------------
__global__ __launch_bounds__(256) void prep_weights(
    const float* __restrict__ Wx1, const float* __restrict__ Wh1,
    const float* __restrict__ Wx2, const float* __restrict__ Wh2,
    __hip_bfloat16* __restrict__ Wb1, __hip_bfloat16* __restrict__ Wb2)
{
    const int per = 9 * NG_ * CC_;                 // 165888 per layer
    int idx = blockIdx.x * 256 + threadIdx.x;
    if (idx >= 2 * per) return;
    const float* Wx = (idx < per) ? Wx1 : Wx2;
    const float* Wh = (idx < per) ? Wh1 : Wh2;
    __hip_bfloat16* Wb = (idx < per) ? Wb1 : Wb2;
    int r  = idx % per;
    int k9 = r / (NG_ * CC_);
    int r2 = r % (NG_ * CC_);
    int n  = r2 / CC_;
    int c  = r2 % CC_;
    float v = (c < C_) ? Wx[((size_t)(k9 * C_) + c) * NG_ + n]
                       : Wh[((size_t)(k9 * C_) + (c - C_)) * NG_ + n];
    Wb[r] = __float2bfloat16(v);
}

// ---------------------------------------------------------------------------
// Pack: write bf16 x-part of the act buffer from the per-timestep input
// (optionally applying inference BatchNorm). At t==0 also zero the h-part
// and the cell state.   src layout: [B,T,HW,48]
// ---------------------------------------------------------------------------
__global__ __launch_bounds__(256) void pack_input(
    const float* __restrict__ src,
    __hip_bfloat16* __restrict__ actX,
    __hip_bfloat16* __restrict__ actH0,       // non-null at t==0: zero h-part
    float* __restrict__ cState,               // non-null at t==0: zero
    const float* __restrict__ gamma, const float* __restrict__ beta,
    const float* __restrict__ mean,  const float* __restrict__ var,
    int t)
{
    int idx = blockIdx.x * 256 + threadIdx.x;  // over PIX_*48
    if (idx >= PIX_ * C_) return;
    int p = idx / C_, c = idx % C_;
    int b = p >> 12, yx = p & (HW_ - 1);
    float v = src[((size_t)(b * T_ + t) * HW_ + yx) * C_ + c];
    if (gamma) {
        float sc = gamma[c] * rsqrtf(var[c] + 1e-3f);
        v = sc * (v - mean[c]) + beta[c];
    }
    actX[(size_t)p * CC_ + c] = __float2bfloat16(v);
    if (cState) cState[idx] = 0.0f;
    if (actH0)  actH0[(size_t)p * CC_ + C_ + c] = __float2bfloat16(0.0f);
}

// ---------------------------------------------------------------------------
// One ConvLSTM timestep. Workgroup: 128 pixels (2 image rows) x 192 gate
// channels; 8 waves, wave w owns M-tile w x all 12 N-tiles so each lane holds
// the full i,f,g,o quadruple in registers. Per 3x3 tap, wave 0 TDM-prefetches
// the next tap's 36 KB weight block into the other LDS buffer.
// ---------------------------------------------------------------------------
__global__ __launch_bounds__(256) void lstm_step(
    const __hip_bfloat16* __restrict__ actIn,   // [PIX][96] bf16
    __hip_bfloat16* __restrict__ actOut,        // h -> [p*96+48+ch]
    const __hip_bfloat16* __restrict__ Wb,      // [9][192][96] bf16
    const float* __restrict__ bias,             // [192]
    float* __restrict__ cState,                 // [PIX][48]
    float* __restrict__ outSeq,                 // [B,T,HW,48]
    const float* __restrict__ resid,            // nullptr or x (same layout)
    int t)
{
    extern __shared__ __align__(16) char smem[];
    __bf16* sAct = (__bf16*)smem;                         // [4][64][96]
    __bf16* sB   = (__bf16*)(smem + ACT_LDS_BYTES);       // [2][192][96]

    const int tid  = threadIdx.x;
    const int lane = tid & 31;
    const int wave = tid >> 5;
    const int tile = blockIdx.x;           // 0..127
    const int b    = tile >> 5;            // image index
    const int y0   = (tile & 31) * 2;      // first of 2 rows in this tile

    // ---- kick off TDM for tap 0 while we stage the act halo -------------
    if (wave == 0)
        tdm_load_tap(ACT_LDS_BYTES, Wb);

    // ---- stage rows y0-1 .. y0+2 of act into LDS (zero OOB rows) --------
    const int rowElems = 64 * CC_;         // 6144 bf16 per row
    for (int i = tid; i < 3072; i += 256) {          // 3072 uint4 total
        int r   = i / 768;                            // 768 uint4 per row
        int off = (i % 768) * 8;                      // bf16 offset in row
        int gy  = y0 - 1 + r;
        uint4 v = make_uint4(0u, 0u, 0u, 0u);
        if (gy >= 0 && gy < 64)
            v = *(const uint4*)&actIn[((size_t)b * HW_ + gy * 64) * CC_ + off];
        *(uint4*)&sAct[r * rowElems + off] = v;
    }
    if (wave == 0)
        __builtin_amdgcn_s_wait_tensorcnt(0);
    __syncthreads();

    // Per-lane A-fragment coordinates (ISA 16-bit A layout: lanes 0-15 hold
    // K in {0..7,16..23}; lanes 16-31 hold K in {8..15,24..31}).
    const int mloc   = wave * 16 + (lane & 15);       // pixel row in tile
    const int yy     = mloc >> 6;                     // 0/1
    const int xx     = mloc & 63;
    const int kgrp   = lane >> 4;                     // 0/1
    const int aoff   = kgrp * 8;                      // A K-base
    const int bkBase = kgrp * 8;                      // B K-base

    f32x8 acc[12] = {};

    for (int k9 = 0; k9 < 9; ++k9) {
        // prefetch next tap's weights into the other LDS buffer
        if (wave == 0 && k9 < 8)
            tdm_load_tap(ACT_LDS_BYTES + (unsigned)(((k9 + 1) & 1) * WTAP_BYTES),
                         Wb + (size_t)(k9 + 1) * WTAP_ELEMS);

        const int dy = k9 / 3 - 1, dx = k9 % 3 - 1;
        const int sy = y0 + yy + dy;
        const int sx = xx + dx;
        const bool ok = (sy >= 0) & (sy < 64) & (sx >= 0) & (sx < 64);
        const int ldsRow  = yy + dy + 1;                       // 0..3
        const int aRowOff = (ldsRow * 64 + sx) * CC_;          // bf16 elems
        const __bf16* bcur = sB + (size_t)(k9 & 1) * WTAP_ELEMS;

        #pragma unroll
        for (int cc = 0; cc < 3; ++cc) {
            const int c0 = cc * 32;
            // ---- A fragment: 4 x ds_load_b64, zeroed for SAME padding
            union { bf16x16 v; uint2 q[4]; } A;
            const int k0 = c0 + aoff;
            if (ok) {
                A.q[0] = *(const uint2*)&sAct[aRowOff + k0 + 0];
                A.q[1] = *(const uint2*)&sAct[aRowOff + k0 + 4];
                A.q[2] = *(const uint2*)&sAct[aRowOff + k0 + 16];
                A.q[3] = *(const uint2*)&sAct[aRowOff + k0 + 20];
            } else {
                A.q[0] = A.q[1] = A.q[2] = A.q[3] = make_uint2(0u, 0u);
            }
            // ---- all 12 B fragments first (clause the ds_loads) ---------
            union { bf16x16 v; uint4 q[2]; } Bf[12];
            #pragma unroll
            for (int nt = 0; nt < 12; ++nt) {
                const __bf16* bp = bcur + (size_t)(nt * 16 + (lane & 15)) * CC_
                                        + c0 + bkBase;
                Bf[nt].q[0] = *(const uint4*)(bp);
                Bf[nt].q[1] = *(const uint4*)(bp + 16);
            }
            #pragma unroll
            for (int nt = 0; nt < 12; ++nt) {
                acc[nt] = __builtin_amdgcn_wmma_f32_16x16x32_bf16(
                    false, A.v, false, Bf[nt].v, (short)0, acc[nt], false, false);
            }
        }

        // hand the double buffer over: everyone done reading buf[k9&1],
        // and the TDM fill of buf[(k9+1)&1] has landed.
        __syncthreads();
        if (wave == 0 && k9 < 8)
            __builtin_amdgcn_s_wait_tensorcnt(0);
        __syncthreads();
    }

    // ---- LSTM pointwise update. N-tiles 0-2 = i, 3-5 = f, 6-8 = g,
    //      9-11 = o: each lane owns the whole quadruple. ------------------
    const int col     = lane & 15;
    const int mAdd    = (lane >> 4) * 8;
    const int pixBase = tile * 128 + wave * 16;

    #pragma unroll
    for (int ft = 0; ft < 3; ++ft) {
        const int ch = ft * 16 + col;
        const float bi = bias[  0 + ch];
        const float bf = bias[ 48 + ch];
        const float bg = bias[ 96 + ch];
        const float bo = bias[144 + ch];
        #pragma unroll
        for (int r = 0; r < 8; ++r) {
            const int p = pixBase + r + mAdd;
            float zi = acc[0 + ft][r] + bi;
            float zf = acc[3 + ft][r] + bf;
            float zg = acc[6 + ft][r] + bg;
            float zo = acc[9 + ft][r] + bo;
            // Keras hard_sigmoid: clip(x/6 + 0.5, 0, 1)
            float si = fminf(fmaxf(zi * (1.0f / 6.0f) + 0.5f, 0.0f), 1.0f);
            float sf = fminf(fmaxf(zf * (1.0f / 6.0f) + 0.5f, 0.0f), 1.0f);
            float so = fminf(fmaxf(zo * (1.0f / 6.0f) + 0.5f, 0.0f), 1.0f);
            float cNew = sf * cState[(size_t)p * C_ + ch] + si * tanhf(zg);
            float h    = so * tanhf(cNew);
            cState[(size_t)p * C_ + ch] = cNew;
            actOut[(size_t)p * CC_ + C_ + ch] = __float2bfloat16(h);
            const int bb = p >> 12, yx = p & (HW_ - 1);
            const size_t oi = ((size_t)(bb * T_ + t) * HW_ + yx) * C_ + ch;
            outSeq[oi] = resid ? (resid[oi] + h) : h;
        }
    }
}

// ---------------------------------------------------------------------------
extern "C" void kernel_launch(void* const* d_in, const int* in_sizes, int n_in,
                              void* d_out, int out_size, void* d_ws, size_t ws_size,
                              hipStream_t stream)
{
    (void)in_sizes; (void)n_in; (void)out_size; (void)ws_size;
    const float* x     = (const float*)d_in[0];
    const float* Wx1   = (const float*)d_in[1];
    const float* Wh1   = (const float*)d_in[2];
    const float* b1    = (const float*)d_in[3];
    const float* Wx2   = (const float*)d_in[4];
    const float* Wh2   = (const float*)d_in[5];
    const float* b2    = (const float*)d_in[6];
    const float* gamma = (const float*)d_in[7];
    const float* beta  = (const float*)d_in[8];
    const float* mean  = (const float*)d_in[9];
    const float* var   = (const float*)d_in[10];
    float* out = (float*)d_out;

    // ---- workspace carve-up (bytes) -------------------------------------
    char* ws = (char*)d_ws;
    const size_t WB_BYTES  = (size_t)9 * NG_ * CC_ * 2;      //   331,776 /layer
    const size_t ACT_BYTES = (size_t)PIX_ * CC_ * 2;         // 3,145,728 /buffer
    const size_t C_BYTES   = (size_t)PIX_ * C_ * 4;          // 3,145,728
    size_t off = 0;
    __hip_bfloat16* Wb1  = (__hip_bfloat16*)(ws + off); off += WB_BYTES;
    __hip_bfloat16* Wb2  = (__hip_bfloat16*)(ws + off); off += WB_BYTES;
    __hip_bfloat16* act0 = (__hip_bfloat16*)(ws + off); off += ACT_BYTES;
    __hip_bfloat16* act1 = (__hip_bfloat16*)(ws + off); off += ACT_BYTES;
    float*          cws  = (float*)(ws + off);          off += C_BYTES;
    float*          h1   = (float*)(ws + off);          // [B,T,HW,48] f32

    prep_weights<<<(2 * 9 * NG_ * CC_ + 255) / 256, 256, 0, stream>>>(
        Wx1, Wh1, Wx2, Wh2, Wb1, Wb2);

    const int packGrid = (PIX_ * C_ + 255) / 256;   // 3072

    // ---- layer 1 --------------------------------------------------------
    for (int t = 0; t < T_; ++t) {
        __hip_bfloat16* aIn  = (t & 1) ? act1 : act0;
        __hip_bfloat16* aOut = (t & 1) ? act0 : act1;
        pack_input<<<packGrid, 256, 0, stream>>>(
            x, aIn, (t == 0) ? aIn : nullptr, (t == 0) ? cws : nullptr,
            nullptr, nullptr, nullptr, nullptr, t);
        lstm_step<<<128, 256, SMEM_BYTES, stream>>>(
            aIn, aOut, Wb1, b1, cws, h1, nullptr, t);
    }

    // ---- BatchNorm folded into pack; layer 2 fuses residual add ---------
    for (int t = 0; t < T_; ++t) {
        __hip_bfloat16* aIn  = (t & 1) ? act1 : act0;
        __hip_bfloat16* aOut = (t & 1) ? act0 : act1;
        pack_input<<<packGrid, 256, 0, stream>>>(
            h1, aIn, (t == 0) ? aIn : nullptr, (t == 0) ? cws : nullptr,
            gamma, beta, mean, var, t);
        lstm_step<<<128, 256, SMEM_BYTES, stream>>>(
            aIn, aOut, Wb2, b2, cws, out, x, t);
    }
}
